// IntraAttention_23398981829190
// MI455X (gfx1250) — matrix operand order repaired and checked by reading, not compiled, not run on hardware
//
#include <hip/hip_runtime.h>
#include <hip/hip_bf16.h>

typedef __attribute__((ext_vector_type(16))) __bf16 v16bf;
typedef __attribute__((ext_vector_type(8)))  __bf16 v8bf;
typedef __attribute__((ext_vector_type(4)))  __bf16 v4bf;
typedef __attribute__((ext_vector_type(8)))  float  v8f;

#define B_    1024
#define L_    100
#define D_    512
#define LP    112     // L padded to 7*16 tiles
#define KATT  128     // attn K padded (112 -> 128) for 32-wide K steps
#define XBROW 128     // x^T staged with 128 zero-padded columns (m index)
#define GROWS 64      // rows of [B*L] per GRU block

// ---------- helpers ----------

__device__ __forceinline__ __bf16 f2bf(float f) {
  unsigned u = __builtin_bit_cast(unsigned, f);
  u += 0x7FFFu + ((u >> 16) & 1u);       // round-to-nearest-even
  unsigned short h = (unsigned short)(u >> 16);
  return __builtin_bit_cast(__bf16, h);
}

__device__ __forceinline__ float wave_max(float v) {
#pragma unroll
  for (int o = 16; o > 0; o >>= 1) v = fmaxf(v, __shfl_xor(v, o, 32));
  return v;
}
__device__ __forceinline__ float wave_sum(float v) {
#pragma unroll
  for (int o = 16; o > 0; o >>= 1) v += __shfl_xor(v, o, 32);
  return v;
}

// A fragment (16x32 bf16): lane half h picks K [16h..16h+15] (contiguous)
__device__ __forceinline__ v16bf load_a_frag(const __bf16* row, int h) {
  v8bf lo = *(const v8bf*)(row + 16 * h);
  v8bf hi = *(const v8bf*)(row + 16 * h + 8);
  v16bf a;
#pragma unroll
  for (int i = 0; i < 8; ++i) { a[i] = lo[i]; a[8 + i] = hi[i]; }
  return a;
}

__device__ __forceinline__ v8f wmma_bf16(v16bf a, v16bf b, v8f c) {
  // D = A(16x32 bf16) * B(32x16 bf16) + C(16x16 f32)
  return __builtin_amdgcn_wmma_f32_16x16x32_bf16(false, a, false, b,
                                                 (short)0, c, false, false);
}

__device__ __forceinline__ float sigmoidf_(float x) {
  return 1.0f / (1.0f + __expf(-x));
}

// ---------- kernel 1: weight convert + transpose  W[K=1024][N=512] -> WT bf16 [N][K] ----------

__global__ void __launch_bounds__(256)
wconv_kernel(const float* __restrict__ Wr, const float* __restrict__ Wu,
             const float* __restrict__ Wo,
             __bf16* __restrict__ WrT, __bf16* __restrict__ WuT,
             __bf16* __restrict__ WoT) {
  const float* src = (blockIdx.y == 0) ? Wr : (blockIdx.y == 1) ? Wu : Wo;
  __bf16* dst = (blockIdx.y == 0) ? WrT : (blockIdx.y == 1) ? WuT : WoT;
  int e = blockIdx.x * 256 + threadIdx.x;         // over 512*1024
  if (e < D_ * 2 * D_) {
    int n = e >> 10;                               // output column
    int k = e & (2 * D_ - 1);                      // contraction index
    dst[e] = f2bf(src[(size_t)k * D_ + n]);
  }
}

// ---------- kernel 2: attention + message (one workgroup per batch) ----------

__global__ void __launch_bounds__(256)
attn_kernel(const float* __restrict__ x, const int* __restrict__ text_len,
            float* __restrict__ msg) {
  extern __shared__ char smem[];
  __bf16* s_xp = (__bf16*)smem;                    // [LP][D_]      xp = x + pe, bf16
  __bf16* s_xt = s_xp + LP * D_;                   // [D_][XBROW]   x^T, bf16, zero-padded m
  float*  s_sc = (float*)(s_xt + (size_t)D_ * XBROW); // [LP][LP]   scores f32
  __bf16* s_at = (__bf16*)(s_sc + LP * LP);        // [LP][KATT]    softmax(attn), bf16

  const int b    = blockIdx.x;
  const int tid  = threadIdx.x;
  const int wave = tid >> 5;
  const int lane = tid & 31;
  const int h    = lane >> 4;      // lane half
  const int r    = lane & 15;      // row/col within 16-tile
  const int len  = text_len[b];
  const float* xb = x + (size_t)b * L_ * D_;

  // ---- stage xp = x + PE (row-major) and x^T (column index = sequence pos) ----
  for (int e = tid * 4; e < XBROW * D_; e += 256 * 4) {
    const int l = e >> 9, d = e & (D_ - 1);
    float4 xv = make_float4(0.f, 0.f, 0.f, 0.f);
    if (l < L_) xv = *(const float4*)(xb + l * D_ + d);
    // transposed x (message-GEMM B operand): s_xt[d][l]
    s_xt[(d + 0) * XBROW + l] = f2bf(xv.x);
    s_xt[(d + 1) * XBROW + l] = f2bf(xv.y);
    s_xt[(d + 2) * XBROW + l] = f2bf(xv.z);
    s_xt[(d + 3) * XBROW + l] = f2bf(xv.w);
    if (l < LP) {
      v4bf xp4;
      if (l < L_) {
        // pe[l,2k]=sin(l*div_k), pe[l,2k+1]=cos(l*div_k), div_k=exp(2k*(-ln1e4/512))
        const float a0 = (float)l * __expf((float)(d & ~1) * -0.017988946039015984f);
        const float a1 = (float)l * __expf((float)((d + 2) & ~1) * -0.017988946039015984f);
        xp4[0] = f2bf(xv.x + __sinf(a0));
        xp4[1] = f2bf(xv.y + __cosf(a0));
        xp4[2] = f2bf(xv.z + __sinf(a1));
        xp4[3] = f2bf(xv.w + __cosf(a1));
      } else {
        xp4[0] = xp4[1] = xp4[2] = xp4[3] = f2bf(0.0f);
      }
      *(v4bf*)(s_xp + l * D_ + d) = xp4;
    }
  }
  __syncthreads();

  // ---- score = xp @ xp^T : 7x7 tiles of 16x16, K = 512, double-buffered ----
  for (int t = wave; t < 49; t += 8) {
    const int mt = t / 7, nt = t % 7;
    const __bf16* arow = s_xp + (mt * 16 + r) * D_;
    const __bf16* brow = s_xp + (nt * 16 + r) * D_ + 16 * h;  // B frag: N = r
    v8f c = {};
    v16bf a0 = load_a_frag(arow, h);
    v16bf b0 = *(const v16bf*)(brow);
#pragma unroll
    for (int kb = 0; kb < D_; kb += 32) {
      v16bf a1 = a0, b1 = b0;
      if (kb + 32 < D_) {                          // prefetch next K-step
        a1 = load_a_frag(arow + kb + 32, h);
        b1 = *(const v16bf*)(brow + kb + 32);
      }
      c = wmma_bf16(a0, b0, c);
      a0 = a1; b0 = b1;
    }
#pragma unroll
    for (int v = 0; v < 8; ++v)
      s_sc[(mt * 16 + 8 * h + v) * LP + nt * 16 + r] = c[v];
  }
  __syncthreads();

  // ---- mask (fill = 1e-10, per reference) + softmax over m in [0,100), bf16 attn ----
  for (int row = wave; row < LP; row += 8) {
    if (row < L_) {
      const bool rv = row < len;
      const float* sr = s_sc + row * LP;
      float vals[4];
      float mx = -3.4e38f;
#pragma unroll
      for (int j = 0; j < 4; ++j) {
        int m = lane + 32 * j;
        float v = -3.4e38f;
        if (m < L_) v = (rv && m < len) ? sr[m] : 1e-10f;
        vals[j] = v;
        mx = fmaxf(mx, v);
      }
      mx = wave_max(mx);
      float sum = 0.0f;
#pragma unroll
      for (int j = 0; j < 4; ++j) {
        int m = lane + 32 * j;
        float e = (m < L_) ? __expf(vals[j] - mx) : 0.0f;
        vals[j] = e;
        sum += e;
      }
      sum = wave_sum(sum);
      const float inv = 1.0f / sum;
#pragma unroll
      for (int j = 0; j < 4; ++j) {
        int m = lane + 32 * j;   // covers 0..127 = KATT
        s_at[row * KATT + m] = f2bf((m < L_) ? vals[j] * inv : 0.0f);
      }
    } else {
      for (int m = lane; m < KATT; m += 32) s_at[row * KATT + m] = f2bf(0.0f);
    }
  }
  __syncthreads();

  // ---- message = attn @ x : 7x32 tiles, K = 128 (pads are zero), x^T gives b128 B loads ----
  float* mb = msg + (size_t)b * L_ * D_;
  for (int t = wave; t < 7 * 32; t += 8) {
    const int mt = t >> 5, nt = t & 31;
    const __bf16* arow = s_at + (mt * 16 + r) * KATT;
    const int dcol = nt * 16 + r;                  // B frag: N = r
    const __bf16* bcol = s_xt + (size_t)dcol * XBROW + 16 * h;  // K contiguous
    v8f c = {};
    v16bf a0 = load_a_frag(arow, h);
    v16bf b0 = *(const v16bf*)(bcol);
#pragma unroll
    for (int kb = 0; kb < KATT; kb += 32) {
      v16bf a1 = a0, b1 = b0;
      if (kb + 32 < KATT) {
        a1 = load_a_frag(arow + kb + 32, h);
        b1 = *(const v16bf*)(bcol + kb + 32);
      }
      c = wmma_bf16(a0, b0, c);
      a0 = a1; b0 = b1;
    }
#pragma unroll
    for (int v = 0; v < 8; ++v) {
      const int rowo = mt * 16 + 8 * h + v;
      if (rowo < L_) mb[rowo * D_ + dcol] = c[v];
    }
  }
}

// ---------- GRU gate GEMM: 8 N-tiles, K=1024, ping-pong B buffers ----------

__device__ __forceinline__ void gate_gemm(const __bf16* __restrict__ a1,  // K 0..511 (row base)
                                          const __bf16* __restrict__ a2,  // K 512..1023 (row base)
                                          const __bf16* __restrict__ WT,
                                          const size_t* __restrict__ wb,  // 8 per-tile bases
                                          int h, v8f acc[8]) {
  v16bf b0[8], b1[8];
#pragma unroll
  for (int j = 0; j < 8; ++j) b0[j] = *(const v16bf*)(WT + wb[j]);
  for (int ks2 = 0; ks2 < 16; ++ks2) {
    const int kbA = ks2 * 64;
    const int kbB = kbA + 32;
#pragma unroll
    for (int j = 0; j < 8; ++j) b1[j] = *(const v16bf*)(WT + wb[j] + kbB);
    {
      const __bf16* ap = (kbA < D_) ? a1 + kbA : a2 + (kbA - D_);
      v16bf a = load_a_frag(ap, h);
#pragma unroll
      for (int j = 0; j < 8; ++j) acc[j] = wmma_bf16(a, b0[j], acc[j]);
    }
    if (ks2 + 1 < 16) {
#pragma unroll
      for (int j = 0; j < 8; ++j) b0[j] = *(const v16bf*)(WT + wb[j] + kbA + 64);
    }
    {
      const __bf16* ap = (kbB < D_) ? a1 + kbB : a2 + (kbB - D_);
      v16bf a = load_a_frag(ap, h);
#pragma unroll
      for (int j = 0; j < 8; ++j) acc[j] = wmma_bf16(a, b1[j], acc[j]);
    }
  }
}

// ---------- kernel 3: fused GRU gates (64 rows of [B*L] per block, 16 waves) ----------

__global__ void __launch_bounds__(512)
gru_kernel(const float* __restrict__ xin, const float* __restrict__ msg,
           const __bf16* __restrict__ WrT, const __bf16* __restrict__ WuT,
           const __bf16* __restrict__ WoT,
           const float* __restrict__ b_r, const float* __restrict__ b_u,
           const float* __restrict__ b_o, float* __restrict__ xout) {
  extern __shared__ char smem[];
  __bf16* s_m  = (__bf16*)smem;             // [GROWS][512] message bf16
  __bf16* s_x  = s_m + GROWS * D_;          // [GROWS][512] x bf16
  __bf16* s_xr = s_x + GROWS * D_;          // [GROWS][512] x*reset bf16

  const int tid  = threadIdx.x;
  const int wave = tid >> 5;
  const int lane = tid & 31;
  const int h    = lane >> 4;
  const int r    = lane & 15;
  const size_t row0 = (size_t)blockIdx.x * GROWS;

  // vectorized staging: f32 -> bf16
  for (int e = tid * 4; e < GROWS * D_; e += 512 * 4) {
    const size_t g = row0 * D_ + e;
    const float4 m4 = *(const float4*)(msg + g);
    const float4 x4 = *(const float4*)(xin + g);
    v4bf mb, xb4;
    mb[0] = f2bf(m4.x); mb[1] = f2bf(m4.y); mb[2] = f2bf(m4.z); mb[3] = f2bf(m4.w);
    xb4[0] = f2bf(x4.x); xb4[1] = f2bf(x4.y); xb4[2] = f2bf(x4.z); xb4[3] = f2bf(x4.w);
    *(v4bf*)(s_m + e) = mb;
    *(v4bf*)(s_x + e) = xb4;
  }
  __syncthreads();

  const int mt  = wave & 3;              // 4 row tiles of 16
  const int nt0 = (wave >> 2) * 8;       // 8 col tiles of 16 per wave
  const __bf16* am = s_m  + (mt * 16 + r) * D_;   // A-frag rows (M = r)
  const __bf16* ax = s_x  + (mt * 16 + r) * D_;
  const __bf16* ar = s_xr + (mt * 16 + r) * D_;

  size_t wb[8];
#pragma unroll
  for (int j = 0; j < 8; ++j)
    wb[j] = (size_t)((nt0 + j) * 16 + r) * (2 * D_) + 16 * h;

  // ---- reset gate: R = sigmoid([msg,x] @ W_r + b_r); stage x*R as bf16 ----
  {
    v8f accR[8];
#pragma unroll
    for (int i = 0; i < 8; ++i) accR[i] = v8f{};
    gate_gemm(am, ax, WrT, wb, h, accR);
#pragma unroll
    for (int j = 0; j < 8; ++j) {
      const int ncol = (nt0 + j) * 16 + r;
      const float br = b_r[ncol];
#pragma unroll
      for (int v = 0; v < 8; ++v) {
        const int rr = mt * 16 + 8 * h + v;
        const float rg = sigmoidf_(accR[j][v] + br);
        const float xv = (float)s_x[rr * D_ + ncol];
        s_xr[rr * D_ + ncol] = f2bf(xv * rg);
      }
    }
  }
  __syncthreads();

  // ---- update gate ----
  v8f accU[8];
#pragma unroll
  for (int i = 0; i < 8; ++i) accU[i] = v8f{};
  gate_gemm(am, ax, WuT, wb, h, accU);

  // ---- candidate: tanh([msg, x*r] @ W_o + b_o) ----
  v8f accC[8];
#pragma unroll
  for (int i = 0; i < 8; ++i) accC[i] = v8f{};
  gate_gemm(am, ar, WoT, wb, h, accC);

  // ---- epilogue: x' = x*(1-u) + tanh(cand)*u ----
#pragma unroll
  for (int j = 0; j < 8; ++j) {
    const int ncol = (nt0 + j) * 16 + r;
    const float bu = b_u[ncol], bo = b_o[ncol];
#pragma unroll
    for (int v = 0; v < 8; ++v) {
      const int rr = mt * 16 + 8 * h + v;
      const size_t g = (row0 + rr) * D_ + ncol;
      const float u = sigmoidf_(accU[j][v] + bu);
      const float c = tanhf(accC[j][v] + bo);
      const float xi = xin[g];
      xout[g] = xi * (1.0f - u) + c * u;
    }
  }
}

// ---------- kernel 4: masked mean pooling ----------

__global__ void __launch_bounds__(256)
pool_kernel(const float* __restrict__ xf, const int* __restrict__ text_len,
            float* __restrict__ emb) {
  const int b = blockIdx.x;
  const int len = text_len[b];
  const float inv = 1.0f / (float)len;
  for (int d = threadIdx.x; d < D_; d += 256) {
    float s = 0.0f;
    for (int l = 0; l < len; ++l) s += xf[((size_t)b * L_ + l) * D_ + d];
    emb[(size_t)b * D_ + d] = s * inv;
  }
}

// ---------- host launcher ----------

extern "C" void kernel_launch(void* const* d_in, const int* in_sizes, int n_in,
                              void* d_out, int out_size, void* d_ws, size_t ws_size,
                              hipStream_t stream) {
  const float* x  = (const float*)d_in[0];
  const int*   tl = (const int*)d_in[1];
  const float* Wr = (const float*)d_in[2];
  const float* br = (const float*)d_in[3];
  const float* Wu = (const float*)d_in[4];
  const float* bu = (const float*)d_in[5];
  const float* Wo = (const float*)d_in[6];
  const float* bo = (const float*)d_in[7];

  const size_t XN = (size_t)B_ * L_ * D_;
  char* ws = (char*)d_ws;
  float*  msg  = (float*)ws;  ws += XN * 4;
  float*  xbuf = (float*)ws;  ws += XN * 4;
  __bf16* WrT  = (__bf16*)ws; ws += (size_t)2 * D_ * D_ * 2;
  __bf16* WuT  = (__bf16*)ws; ws += (size_t)2 * D_ * D_ * 2;
  __bf16* WoT  = (__bf16*)ws;

  float* x_final = (float*)d_out;       // [B, L, D]
  float* emb     = x_final + XN;        // [B, D]

  wconv_kernel<<<dim3((D_ * 2 * D_) / 256, 3), 256, 0, stream>>>(Wr, Wu, Wo, WrT, WuT, WoT);

  const size_t SM_A = (size_t)LP * D_ * 2 + (size_t)D_ * XBROW * 2 +
                      (size_t)LP * LP * 4 + (size_t)LP * KATT * 2;   // ~317 KB
  const size_t SM_G = (size_t)3 * GROWS * D_ * 2;                     // 192 KB

  const float* xi = x;
  float* xo = xbuf;
  for (int layer = 0; layer < 2; ++layer) {
    attn_kernel<<<B_, 256, SM_A, stream>>>(xi, tl, msg);
    gru_kernel<<<(B_ * L_) / GROWS, 512, SM_G, stream>>>(xi, msg, WrT, WuT, WoT,
                                                         br, bu, bo, xo);
    xi = xo;
    xo = x_final;
  }
  pool_kernel<<<B_, 256, 0, stream>>>(x_final, tl, emb);
}
